// compute_probs_50491635532315
// MI455X (gfx1250) — compile-verified, matrix operand order checked by reading, not run on hardware
//
#include <hip/hip_runtime.h>
#include <hip/hip_bf16.h>

typedef __attribute__((ext_vector_type(16))) _Float16 v16h;
typedef __attribute__((ext_vector_type(8)))  float    v8f;

#define BB   8
#define CC   512
#define HH   80
#define WWD  80
#define HWD  (HH*WWD)      // 6400
#define NPIX (BB*HWD)      // 51200
#define NREF 100
#define HR   40
#define WR   40
#define NCLS 20

// ---------------------------------------------------------------------------
// K1: per-pixel inverse L2 norm over 512 channels (coalesced across pixels)
// ---------------------------------------------------------------------------
__global__ void invnorm_kernel(const float* __restrict__ src, float* __restrict__ inv) {
    int p = blockIdx.x * blockDim.x + threadIdx.x;
    if (p >= NPIX) return;
    int b = p / HWD, rem = p % HWD;
    const float* base = src + (size_t)b * CC * HWD + rem;
    float acc = 0.f;
    #pragma unroll 8
    for (int c = 0; c < CC; ++c) { float v = base[(size_t)c * HWD]; acc += v * v; }
    inv[p] = 1.0f / fmaxf(sqrtf(acc), 1e-10f);
}

// ---------------------------------------------------------------------------
// K2: gather ref vectors, segment-mean into 20 protos, row-normalize.
// Emits: s[c] = sum_v P_norm[v][c] (fp32 path for argmin), and f16 B-matrix
// fragments pre-swizzled into the CDNA5 WMMA 16-bit B VGPR layout:
//   lane L: N = L&15, kg = (L>>4)*8; element i: K = kb*32 + i + kg + (i>=8?8:0)
// ---------------------------------------------------------------------------
__global__ void __launch_bounds__(512)
protos_kernel(const float* __restrict__ ref_c,
              const int*   __restrict__ coord,
              const int*   __restrict__ cls,
              float*       __restrict__ s_out,
              _Float16*    __restrict__ bfrag) {
    __shared__ float protos[NCLS][CC];   // 40 KB of the 320 KB WGP LDS
    __shared__ float invn[NCLS];
    __shared__ float cnt[NCLS];
    int tid = threadIdx.x;               // 512 threads; tid == channel
    if (tid < NCLS) {
        int c = 0;
        for (int n = 0; n < NREF; ++n) if (cls[n] == tid) ++c;
        cnt[tid] = fmaxf((float)c, 1.0f);
    }
    __syncthreads();

    float acc[NCLS];
    #pragma unroll
    for (int v = 0; v < NCLS; ++v) acc[v] = 0.f;
    for (int n = 0; n < NREF; ++n) {
        int x  = coord[2 * n + 0];
        int y  = coord[2 * n + 1];
        int cv = cls[n];
        float val = ref_c[(((size_t)n * CC + tid) * HR + y) * WR + x];
        #pragma unroll
        for (int v = 0; v < NCLS; ++v) acc[v] += (cv == v) ? val : 0.f;
    }
    #pragma unroll
    for (int v = 0; v < NCLS; ++v) protos[v][tid] = acc[v] / cnt[v];
    __syncthreads();

    if (tid < NCLS) {
        float sq = 0.f;
        for (int c = 0; c < CC; ++c) { float p = protos[tid][c]; sq += p * p; }
        invn[tid] = 1.0f / fmaxf(sqrtf(sq), 1e-10f);
    }
    __syncthreads();

    float sv = 0.f;
    #pragma unroll
    for (int v = 0; v < NCLS; ++v) sv += protos[v][tid] * invn[v];
    s_out[tid] = sv;

    // B fragments: 2 N-tiles x 16 k-blocks x 32 lanes x 16 halves
    for (int e = tid; e < 2 * 16 * 32 * 16; e += 512) {
        int i    = e & 15;
        int lane = (e >> 4) & 31;
        int kb   = (e >> 9) & 15;
        int tile = e >> 13;
        int n    = tile * 16 + (lane & 15);
        int kg   = (lane >> 4) * 8;
        int k    = kb * 32 + i + kg + ((i >= 8) ? 8 : 0);
        float val = (n < NCLS) ? protos[n][k] * invn[n] : 0.f;
        bfrag[e] = (_Float16)val;
    }
}

// ---------------------------------------------------------------------------
// K3: WMMA GEMM. Each wave: 16 pixels x 32 protos (20 valid), K=512 via
// 16 steps of v_wmma_f32_16x16x32_f16 per N-tile. A fragments follow the
// ISA 16-bit A 16x32 layout (lane = M, kg split). prob_C goes straight
// into d_out channels 1..20; flat[] accumulated on an fp32 side path.
// ---------------------------------------------------------------------------
__global__ void __launch_bounds__(256)
gemm_kernel(const float*    __restrict__ codeC,
            const float*    __restrict__ invC,
            const _Float16* __restrict__ bfrag,
            const float*    __restrict__ svec,
            float*          __restrict__ out,
            float*          __restrict__ flat) {
    int tid  = threadIdx.x;
    int wave = tid >> 5;
    int lane = tid & 31;
    int p0   = blockIdx.x * 128 + wave * 16;   // 16 pixels per wave, row-aligned
    int m    = lane & 15;
    int kg   = (lane >> 4) * 8;
    int pix  = p0 + m;
    int b    = pix / HWD;
    int rem  = pix % HWD;
    const float* aBase = codeC + (size_t)b * CC * HWD + rem;
    float inv = invC[pix];

    v8f c0 = {}; v8f c1 = {};
    float flatacc = 0.f;
    union { v16h v; _Float16 e[16]; } a;
    const v16h* bf = (const v16h*)bfrag;

    for (int kb = 0; kb < 16; ++kb) {
        #pragma unroll
        for (int i = 0; i < 16; ++i) {
            int k   = kb * 32 + i + kg + ((i >= 8) ? 8 : 0);
            float v = aBase[(size_t)k * HWD];
            float nv = v * inv;
            flatacc += nv * svec[k];          // fp32 path for argmin
            a.e[i] = (_Float16)nv;
        }
        v16h b0 = bf[(0 * 16 + kb) * 32 + lane];
        v16h b1 = bf[(1 * 16 + kb) * 32 + lane];
        c0 = __builtin_amdgcn_wmma_f32_16x16x32_f16(false, a.v, false, b0,
                                                    (short)0, c0, false, false);
        c1 = __builtin_amdgcn_wmma_f32_16x16x32_f16(false, a.v, false, b1,
                                                    (short)0, c1, false, false);
    }

    // flat[pixel] = sum over both channel halves (lane m + lane m^16)
    float tot = flatacc + __shfl_xor(flatacc, 16, 32);
    if (lane < 16) flat[p0 + lane] = tot;

    // D layout: lanes 0-15 -> N=lane, M=r; lanes 16-31 -> N=lane-16, M=r+8
    int n0   = lane & 15;
    int mOff = (lane >> 4) * 8;
    int bb   = p0 / HWD;
    int rem0 = p0 % HWD;                       // same (b,h) for whole wave
    float* outBase = out + (size_t)bb * 21 * HWD + rem0;
    #pragma unroll
    for (int r = 0; r < 8; ++r) {
        int mm = mOff + r;
        outBase[(size_t)(1 + n0) * HWD + mm] = c0[r];
        if (n0 + 16 < NCLS)
            outBase[(size_t)(1 + n0 + 16) * HWD + mm] = c1[r];
    }
}

// ---------------------------------------------------------------------------
// K4: per-batch argmin over flat (first-occurrence tie-break, like jnp.argmin)
// ---------------------------------------------------------------------------
__global__ void argmin_kernel(const float* __restrict__ flat, int* __restrict__ idxOut) {
    __shared__ float sv[256];
    __shared__ int   si[256];
    int b = blockIdx.x, tid = threadIdx.x;
    const float* f = flat + b * HWD;
    float bv = 3.4e38f; int bi = 0;
    for (int i = tid; i < HWD; i += 256) {
        float v = f[i];
        if (v < bv) { bv = v; bi = i; }
    }
    sv[tid] = bv; si[tid] = bi;
    __syncthreads();
    for (int s = 128; s > 0; s >>= 1) {
        if (tid < s) {
            float ov = sv[tid + s]; int oi = si[tid + s];
            if (ov < sv[tid] || (ov == sv[tid] && oi < si[tid])) { sv[tid] = ov; si[tid] = oi; }
        }
        __syncthreads();
    }
    if (tid == 0) idxOut[b] = si[0];
}

// ---------------------------------------------------------------------------
// K5: extract the normalized code_BG vector at the argmin pixel per batch
// ---------------------------------------------------------------------------
__global__ void minvec_kernel(const float* __restrict__ codeBG, const float* __restrict__ invBG,
                              const int* __restrict__ idx, float* __restrict__ minvec) {
    int e = blockIdx.x * blockDim.x + threadIdx.x;
    if (e >= BB * CC) return;
    int b = e >> 9, c = e & 511;
    int p = idx[b];
    minvec[e] = codeBG[((size_t)b * CC + c) * HWD + p] * invBG[b * HWD + p];
}

// ---------------------------------------------------------------------------
// K6: prob_BG dot + softmax over the 21 channels (in/out of d_out)
// ---------------------------------------------------------------------------
__global__ void bgsoftmax_kernel(const float* __restrict__ codeBG, const float* __restrict__ invBG,
                                 const float* __restrict__ minvec, float* __restrict__ out) {
    int p = blockIdx.x * blockDim.x + threadIdx.x;
    if (p >= NPIX) return;
    int b = p / HWD, rem = p % HWD;
    const float* base = codeBG + (size_t)b * CC * HWD + rem;
    const float* mv   = minvec + b * CC;
    float acc = 0.f;
    #pragma unroll 8
    for (int c = 0; c < CC; ++c) acc += base[(size_t)c * HWD] * mv[c];
    float probBG = acc * invBG[p];

    float* o = out + (size_t)b * 21 * HWD + rem;
    float vals[21];
    vals[0] = probBG;
    float mx = probBG;
    #pragma unroll
    for (int n = 0; n < NCLS; ++n) { float v = o[(size_t)(1 + n) * HWD]; vals[1 + n] = v; mx = fmaxf(mx, v); }
    float sum = 0.f;
    #pragma unroll
    for (int j = 0; j < 21; ++j) { float e = __expf(vals[j] - mx); vals[j] = e; sum += e; }
    float is = 1.0f / sum;
    #pragma unroll
    for (int j = 0; j < 21; ++j) o[(size_t)j * HWD] = vals[j] * is;
}

// ---------------------------------------------------------------------------
extern "C" void kernel_launch(void* const* d_in, const int* in_sizes, int n_in,
                              void* d_out, int out_size, void* d_ws, size_t ws_size,
                              hipStream_t stream) {
    (void)in_sizes; (void)n_in; (void)out_size; (void)ws_size;
    const float* code_C  = (const float*)d_in[0];
    const float* code_BG = (const float*)d_in[1];
    const float* ref_c   = (const float*)d_in[2];
    /* d_in[3] (ref_code_bg) is unused by the reference */
    const int* coord = (const int*)d_in[4];
    const int* cls   = (const int*)d_in[5];
    float* out = (float*)d_out;

    char* ws = (char*)d_ws;
    size_t off = 0;
    auto alloc = [&](size_t bytes) -> void* {
        void* p = ws + off;
        off = (off + bytes + 255) & ~(size_t)255;
        return p;
    };
    float*    invC   = (float*)alloc((size_t)NPIX * sizeof(float));
    float*    invBG  = (float*)alloc((size_t)NPIX * sizeof(float));
    float*    flat   = (float*)alloc((size_t)NPIX * sizeof(float));
    float*    svec   = (float*)alloc((size_t)CC * sizeof(float));
    int*      idx    = (int*)alloc((size_t)BB * sizeof(int));
    float*    minvec = (float*)alloc((size_t)BB * CC * sizeof(float));
    _Float16* bfrag  = (_Float16*)alloc((size_t)2 * 16 * 32 * 16 * sizeof(_Float16));

    invnorm_kernel<<<NPIX / 256, 256, 0, stream>>>(code_C, invC);
    invnorm_kernel<<<NPIX / 256, 256, 0, stream>>>(code_BG, invBG);
    protos_kernel<<<1, 512, 0, stream>>>(ref_c, coord, cls, svec, bfrag);
    gemm_kernel<<<NPIX / 128, 256, 0, stream>>>(code_C, invC, bfrag, svec, out, flat);
    argmin_kernel<<<BB, 256, 0, stream>>>(flat, idx);
    minvec_kernel<<<(BB * CC) / 256, 256, 0, stream>>>(code_BG, invBG, idx, minvec);
    bgsoftmax_kernel<<<NPIX / 256, 256, 0, stream>>>(code_BG, invBG, minvec, out);
}
